// GraphConvolutionalNetwork_15281493639201
// MI455X (gfx1250) — compile-verified, matrix operand order checked by reading, not run on hardware
//
#include <hip/hip_runtime.h>
#include <hip/hip_bf16.h>

typedef __attribute__((ext_vector_type(16))) __bf16 v16bf;
typedef __attribute__((ext_vector_type(8)))  float  v8f;

#define D_INHID 256
#define D_OUTF  128
#define BN_EPS  1e-5f

__device__ __forceinline__ unsigned short f2bf(float f) {
  unsigned u = __float_as_uint(f);
  u += 0x7fffu + ((u >> 16) & 1u);           // round-to-nearest-even
  return (unsigned short)(u >> 16);
}

// ---------------- degree / normalization ----------------
__global__ void k_deg_init(float* deg, int n) {
  int i = blockIdx.x * blockDim.x + threadIdx.x;
  if (i < n) deg[i] = 1.0f;                   // self-loop contributes 1
}

__global__ void k_deg_edges(const int* __restrict__ dst, float* deg, int e) {
  int i = blockIdx.x * blockDim.x + threadIdx.x;
  if (i < e)
    __hip_atomic_fetch_add(&deg[dst[i]], 1.0f, __ATOMIC_RELAXED,
                           __HIP_MEMORY_SCOPE_AGENT);
}

__global__ void k_dinv(float* deg, int n) {
  int i = blockIdx.x * blockDim.x + threadIdx.x;
  if (i < n) deg[i] = rsqrtf(deg[i]);
}

// ---------------- f32 -> bf16 convert, zero-filling M padding rows ----------------
__global__ void k_f32_to_bf16_pad(const float* __restrict__ in,
                                  unsigned short* __restrict__ out,
                                  int n, int ntot) {
  int i = blockIdx.x * blockDim.x + threadIdx.x;
  if (i >= ntot) return;
  out[i] = (i < n) ? f2bf(in[i]) : (unsigned short)0;
}

// ---------------- pack W (K x Dout, row-major f32) into WMMA B-fragment order --------
// Tile (nt, kt) covers cols nt*16..+15, rows kt*32..+31 of W.
// Fragment: 32 lanes x 16 bf16, lane-contiguous (32B per lane).
// lane<16: n = nt*16+lane, holds K = kt*32 + {0..7, 16..23}
// lane>=16: n = nt*16+lane-16, holds K = kt*32 + {8..15, 24..31}
__global__ void k_pack_w(const float* __restrict__ W, unsigned short* __restrict__ Wpk,
                         int KT, int NT, int Dout) {
  int t = blockIdx.x * blockDim.x + threadIdx.x;
  int total = KT * NT * 512;
  if (t >= total) return;
  int j    = t & 15;
  int lane = (t >> 4) & 31;
  int rem  = t >> 9;
  int kt   = rem % KT;
  int nt   = rem / KT;
  int n  = nt * 16 + (lane & 15);
  int kh = lane >> 4;
  int k  = kt * 32 + kh * 8 + ((j < 8) ? j : (8 + j));   // j>=8 -> +16 window
  Wpk[t] = f2bf(W[(size_t)k * Dout + n]);
}

// ---------------- WMMA GEMM: C[Mpad x Dout] = A[Mpad x K=KT*32](bf16) * Bpk ----------
// One wave computes a 32(M) x 64(N) strip: 8 accumulators; each B fragment feeds
// two WMMAs. Mpad is a multiple of 32 (rows padded with zeros) so the body is
// branch-free; the only guard is a scalar (readfirstlane) early exit, keeping
// EXEC all-1s through every WMMA as the ISA requires. KT compile-time ->
// fully unrolled K loop, all loads use immediate offsets.
template <int KT>
__global__ void k_gemm_wmma(const __bf16* __restrict__ A,
                            const __bf16* __restrict__ Bpk,
                            float* __restrict__ C,
                            int Mpad, int Dout) {
  const int K = KT * 32;
  int lane = threadIdx.x & 31;
  int wid  = __builtin_amdgcn_readfirstlane(
      (int)((blockIdx.x * blockDim.x + threadIdx.x) >> 5));
  const int nWaveCols = Dout >> 6;
  int mt = wid / nWaveCols;
  int nq = wid - mt * nWaveCols;
  int m0 = mt * 32;
  if (m0 >= Mpad) return;                    // scalar branch (SGPR compare)

  const int row = lane & 15;
  const int kh  = lane >> 4;

  v8f z = {};
  v8f acc0[4] = {z, z, z, z};
  v8f acc1[4] = {z, z, z, z};

  const __bf16* a0 = A + (size_t)(m0 + row) * K + kh * 8;
  const __bf16* a1 = a0 + (size_t)16 * K;
  const __bf16* bps[4];
#pragma unroll
  for (int j = 0; j < 4; ++j)
    bps[j] = Bpk + (size_t)((nq * 4 + j) * KT) * 512 + lane * 16;

#pragma unroll
  for (int kt = 0; kt < KT; ++kt) {
    union { uint4 q[2]; v16bf v; } a, a2;
    a.q[0]  = *(const uint4*)(a0 + kt * 32);        // K: kh*8 + 0..7
    a.q[1]  = *(const uint4*)(a0 + kt * 32 + 16);   // K: kh*8 + 16..23
    a2.q[0] = *(const uint4*)(a1 + kt * 32);
    a2.q[1] = *(const uint4*)(a1 + kt * 32 + 16);
#pragma unroll
    for (int j = 0; j < 4; ++j) {
      union { uint4 q[2]; v16bf v; } b;
      b.q[0] = *(const uint4*)(bps[j] + kt * 512);
      b.q[1] = *(const uint4*)(bps[j] + kt * 512 + 8);
      acc0[j] = __builtin_amdgcn_wmma_f32_16x16x32_bf16(
          false, a.v, false, b.v, (short)0, acc0[j], false, false);
      acc1[j] = __builtin_amdgcn_wmma_f32_16x16x32_bf16(
          false, a2.v, false, b.v, (short)0, acc1[j], false, false);
    }
  }

  // C layout: lane 0-15 -> M=vgpr, N=lane ; lane 16-31 -> M=8+vgpr, N=lane-16
  float* c0 = C + (size_t)(m0 + kh * 8) * Dout + nq * 64 + row;
  float* c1 = c0 + (size_t)16 * Dout;
#pragma unroll
  for (int j = 0; j < 4; ++j)
#pragma unroll
    for (int v = 0; v < 8; ++v) {
      c0[(size_t)v * Dout + j * 16] = acc0[j][v];
      c1[(size_t)v * Dout + j * 16] = acc1[j][v];
    }
}

// ---------------- self-loop init: AGG = dinv^2 * H ----------------
__global__ void k_selfloop(const float* __restrict__ H, const float* __restrict__ dinv,
                           float* __restrict__ AGG, int D, int total) {
  int idx = blockIdx.x * blockDim.x + threadIdx.x;
  if (idx >= total) return;
  int i = idx / D;
  float w = dinv[i];
  AGG[idx] = w * w * H[idx];
}

// ---------------- edge scatter: AGG[dst] += dinv[s]*dinv[d] * H[src] -------------
// One wave per edge; each lane handles D/32 contiguous channels (coalesced).
__global__ void k_edge_agg(const int* __restrict__ src, const int* __restrict__ dst,
                           const float* __restrict__ dinv,
                           const float* __restrict__ H, float* __restrict__ AGG,
                           int E, int D) {
  int gid  = blockIdx.x * blockDim.x + threadIdx.x;
  int e    = gid >> 5;
  if (e >= E) return;
  int lane = gid & 31;
  int s = src[e], d = dst[e];
  float w = dinv[s] * dinv[d];
  int ch = D >> 5;
  const float* hs = H   + (size_t)s * D + lane * ch;
  float*       ad = AGG + (size_t)d * D + lane * ch;
#pragma unroll 4
  for (int j = 0; j < ch; ++j)
    __hip_atomic_fetch_add(&ad[j], w * hs[j], __ATOMIC_RELAXED,
                           __HIP_MEMORY_SCOPE_AGENT);
}

// ---------------- bias + BN(eval) + ReLU -> bf16 (next layer input) --------------
__global__ void k_bn_relu(const float* __restrict__ AGG,
                          const float* __restrict__ b, const float* __restrict__ g,
                          const float* __restrict__ be, const float* __restrict__ m,
                          const float* __restrict__ v,
                          unsigned short* __restrict__ out, int D, int total) {
  int idx = blockIdx.x * blockDim.x + threadIdx.x;
  if (idx >= total) return;
  int c = idx & (D - 1);
  float val = (AGG[idx] + b[c] - m[c]) * (g[c] * rsqrtf(v[c] + BN_EPS)) + be[c];
  out[idx] = f2bf(fmaxf(val, 0.0f));
}

// ---------------- final bias add -> f32 output ----------------
__global__ void k_bias_out(const float* __restrict__ AGG, const float* __restrict__ b,
                           float* __restrict__ out, int D, int total) {
  int idx = blockIdx.x * blockDim.x + threadIdx.x;
  if (idx >= total) return;
  out[idx] = AGG[idx] + b[idx & (D - 1)];
}

extern "C" void kernel_launch(void* const* d_in, const int* in_sizes, int n_in,
                              void* d_out, int out_size, void* d_ws, size_t ws_size,
                              hipStream_t stream) {
  const float* x   = (const float*)d_in[0];
  const int*   ei  = (const int*)d_in[1];
  const float* W1  = (const float*)d_in[2];
  const float* b1  = (const float*)d_in[3];
  const float* g1  = (const float*)d_in[4];
  const float* be1 = (const float*)d_in[5];
  const float* m1  = (const float*)d_in[6];
  const float* v1  = (const float*)d_in[7];
  const float* W2  = (const float*)d_in[8];
  const float* b2  = (const float*)d_in[9];
  const float* g2  = (const float*)d_in[10];
  const float* be2 = (const float*)d_in[11];
  const float* m2  = (const float*)d_in[12];
  const float* v2  = (const float*)d_in[13];
  const float* W3  = (const float*)d_in[14];
  const float* b3  = (const float*)d_in[15];
  float* out = (float*)d_out;

  const int N    = in_sizes[0] / D_INHID;   // 50000
  const int Npad = (N + 31) & ~31;          // 50016: GEMM rows padded to 32
  const int E    = in_sizes[1] / 2;         // 800000
  const int* srcI = ei;
  const int* dstI = ei + E;

  // ---- workspace carve (256B aligned) ----
  char* ws = (char*)d_ws;
  size_t off = 0;
  auto carve = [&](size_t bytes) -> void* {
    void* p = ws + off;
    off = (off + bytes + 255) & ~(size_t)255;
    return p;
  };
  float*          dinv = (float*)carve((size_t)N * 4);
  unsigned short* w1pk = (unsigned short*)carve((size_t)16 * 8 * 512 * 2);
  unsigned short* w2pk = (unsigned short*)carve((size_t)16 * 8 * 512 * 2);
  unsigned short* w3pk = (unsigned short*)carve((size_t)8  * 8 * 512 * 2);
  unsigned short* Xb   = (unsigned short*)carve((size_t)Npad * D_INHID * 2);
  float*          H    = (float*)carve((size_t)Npad * D_INHID * 4);
  float*          AGG  = (float*)carve((size_t)N * D_INHID * 4);

  const int TB = 256;
  auto blk = [](long long n) { return (unsigned)((n + 255) / 256); };

  // normalization
  k_deg_init <<<blk(N), TB, 0, stream>>>(dinv, N);
  k_deg_edges<<<blk(E), TB, 0, stream>>>(dstI, dinv, E);
  k_dinv     <<<blk(N), TB, 0, stream>>>(dinv, N);

  // input -> bf16 (pad rows zeroed every call), pack weights
  k_f32_to_bf16_pad<<<blk((long long)Npad * D_INHID), TB, 0, stream>>>(
      x, Xb, N * D_INHID, Npad * D_INHID);
  k_pack_w<<<blk(8 * 16 * 512), TB, 0, stream>>>(W1, w1pk, 8, 16, 256);
  k_pack_w<<<blk(8 * 16 * 512), TB, 0, stream>>>(W2, w2pk, 8, 16, 256);
  k_pack_w<<<blk(8 * 8  * 512), TB, 0, stream>>>(W3, w3pk, 8, 8, 128);

  const int Mt32 = Npad / 32;                    // 1563 full 32-row strips
  long long ndH = (long long)N * D_INHID;
  long long ndO = (long long)N * D_OUTF;

  // ---- layer 1 ----
  {
    long long waves = (long long)Mt32 * (256 >> 6);
    k_gemm_wmma<8><<<blk(waves * 32), TB, 0, stream>>>(
        (const __bf16*)Xb, (const __bf16*)w1pk, H, Npad, 256);
    k_selfloop<<<blk(ndH), TB, 0, stream>>>(H, dinv, AGG, 256, (int)ndH);
    k_edge_agg<<<blk((long long)E * 32), TB, 0, stream>>>(srcI, dstI, dinv, H, AGG, E, 256);
    k_bn_relu <<<blk(ndH), TB, 0, stream>>>(AGG, b1, g1, be1, m1, v1, Xb, 256, (int)ndH);
  }
  // ---- layer 2 ----
  {
    long long waves = (long long)Mt32 * (256 >> 6);
    k_gemm_wmma<8><<<blk(waves * 32), TB, 0, stream>>>(
        (const __bf16*)Xb, (const __bf16*)w2pk, H, Npad, 256);
    k_selfloop<<<blk(ndH), TB, 0, stream>>>(H, dinv, AGG, 256, (int)ndH);
    k_edge_agg<<<blk((long long)E * 32), TB, 0, stream>>>(srcI, dstI, dinv, H, AGG, E, 256);
    k_bn_relu <<<blk(ndH), TB, 0, stream>>>(AGG, b2, g2, be2, m2, v2, Xb, 256, (int)ndH);
  }
  // ---- layer 3 ----
  {
    long long waves = (long long)Mt32 * (128 >> 6);
    k_gemm_wmma<8><<<blk(waves * 32), TB, 0, stream>>>(
        (const __bf16*)Xb, (const __bf16*)w3pk, H, Npad, 128);
    k_selfloop<<<blk(ndO), TB, 0, stream>>>(H, dinv, AGG, 128, (int)ndO);
    k_edge_agg<<<blk((long long)E * 32), TB, 0, stream>>>(srcI, dstI, dinv, H, AGG, E, 128);
    k_bias_out<<<blk(ndO), TB, 0, stream>>>(AGG, b3, out, 128, (int)ndO);
  }
}
